// Attention_72155450573465
// MI455X (gfx1250) — matrix operand order repaired and checked by reading, not compiled, not run on hardware
//
#include <hip/hip_runtime.h>

// Problem sizes (fixed by the reference)
#define B_  32
#define S_  2048
#define D_  1024
#define KT_ 32   // D / 32  (K-tiles of 32 for bf16 WMMA)
#define NT_ 64   // D / 16  (N-tiles of 16)

typedef __attribute__((ext_vector_type(16))) __bf16 v16bf;
typedef __attribute__((ext_vector_type(8)))  float  v8f;

// ---------------------------------------------------------------------------
// Kernel 1: convert W_h (D x D, fp32 row-major) into bf16 WMMA B-fragments.
// B operand layout (32x16 KxN) for V_WMMA_F32_16X16X32_BF16:
//   lane l (0..31): column N = nt*16 + (l & 15)
//                   K values  = kt*32 + (l>>4)*8 + {0..7, 16..23}  (e = 0..7, 8..15)
// Flat index: frag[(((kt*NT_ + nt)*32) + lane)*16 + e]
// => one contiguous 32-byte load per lane per fragment in the hot loop.
// ---------------------------------------------------------------------------
__global__ __launch_bounds__(256) void swizzle_wh_kernel(
    const float* __restrict__ Wh, __bf16* __restrict__ frag)
{
    int t    = blockIdx.x * 256 + threadIdx.x;   // 0 .. 1048575
    int e    = t & 15;
    int lane = (t >> 4) & 31;
    int tile = t >> 9;                           // kt*NT_ + nt
    int nt   = tile & (NT_ - 1);
    int kt   = tile >> 6;
    int k    = (kt << 5) + ((lane >> 4) << 3) + (e & 7) + ((e >> 3) << 4);
    int n    = (nt << 4) + (lane & 15);
    frag[t]  = (__bf16)Wh[k * D_ + n];
}

// ---------------------------------------------------------------------------
// Kernel 2: ws2[b,d] = (cat(state_h,state_c)[b] . W_s[:,d]) + b_s + b_h + b_cov
// ---------------------------------------------------------------------------
__global__ __launch_bounds__(64) void wstate_kernel(
    const float* __restrict__ sh, const float* __restrict__ sc,
    const float* __restrict__ Ws, const float* __restrict__ bs,
    const float* __restrict__ bh, const float* __restrict__ bcov,
    float* __restrict__ ws2)
{
    int b     = blockIdx.x >> 4;          // 32 batches
    int chunk = blockIdx.x & 15;          // 16 chunks of 64 columns
    int d     = (chunk << 6) + threadIdx.x;
    float acc = bs[d] + bh[d] + bcov[d];
    const float* h = sh + b * D_;
    const float* c = sc + b * D_;
    for (int k = 0; k < D_; ++k) acc += h[k] * Ws[k * D_ + d];
    for (int k = 0; k < D_; ++k) acc += c[k] * Ws[(k + D_) * D_ + d];
    ws2[b * D_ + d] = acc;
}

// ---------------------------------------------------------------------------
// Kernel 3: inv_covsum[b] = 1 / sum_s (coverage[b,s] + 1e-3)
// ---------------------------------------------------------------------------
__global__ __launch_bounds__(256) void covsum_kernel(
    const float* __restrict__ cov, float* __restrict__ inv)
{
    int b = blockIdx.x, tid = threadIdx.x;
    float s = 0.f;
    for (int i = tid; i < S_; i += 256) s += cov[b * S_ + i] + 0.001f;
    __shared__ float red[256];
    red[tid] = s; __syncthreads();
    for (int st = 128; st > 0; st >>= 1) {
        if (tid < st) red[tid] += red[tid + st];
        __syncthreads();
    }
    if (tid == 0) inv[b] = 1.0f / red[0];
}

// ---------------------------------------------------------------------------
// Helper: load 16 fp32 context values in A-fragment order and convert to bf16.
// A fragment (16x32 bf16): lane l holds row M=(l&15),
//   K = kt*32 + (l>>4)*8 + {0..7, 16..23}  (elements 0..7, 8..15).
// ---------------------------------------------------------------------------
__device__ __forceinline__ v16bf load_a_frag(const float* __restrict__ ap)
{
    float4 f0 = *(const float4*)(ap);
    float4 f1 = *(const float4*)(ap + 4);
    float4 f2 = *(const float4*)(ap + 16);
    float4 f3 = *(const float4*)(ap + 20);
    v16bf a;
    a[0]  = (__bf16)f0.x; a[1]  = (__bf16)f0.y; a[2]  = (__bf16)f0.z; a[3]  = (__bf16)f0.w;
    a[4]  = (__bf16)f1.x; a[5]  = (__bf16)f1.y; a[6]  = (__bf16)f1.z; a[7]  = (__bf16)f1.w;
    a[8]  = (__bf16)f2.x; a[9]  = (__bf16)f2.y; a[10] = (__bf16)f2.z; a[11] = (__bf16)f2.w;
    a[12] = (__bf16)f3.x; a[13] = (__bf16)f3.y; a[14] = (__bf16)f3.z; a[15] = (__bf16)f3.w;
    return a;
}

// ---------------------------------------------------------------------------
// Kernel 4 (main): per 32-row M-block of (b,s):
//   e[b,s] = v_b + sum_n v_w[n] * tanh( (context@W_h)[s,n] + ws2[b,n]
//                                       + covn[b,s]*W_cov[n] )
// 8 waves/WG; wave w owns N in [w*128, w*128+128).
// Two 16-row A-tiles per wave share every B fragment (halves L2 B traffic:
// 4096 -> 2048 WG reads of the 2MB W_h fragment buffer).
// Accums: 2 x 8 tiles x v8f = 128 VGPRs/wave.
// ---------------------------------------------------------------------------
__global__ __launch_bounds__(256) void attn_e_kernel(
    const float*  __restrict__ context,
    const float*  __restrict__ coverage,
    const __bf16* __restrict__ whf,     // swizzled W_h bf16 fragments
    const float*  __restrict__ ws2,     // w_state + b_h + b_cov
    const float*  __restrict__ invcov,
    const float*  __restrict__ Wcov,    // (D)
    const float*  __restrict__ vw,      // (D)
    const float*  __restrict__ vb,      // (1)
    float*        __restrict__ e_out)   // (B,S)
{
    const int tid   = threadIdx.x;
    const int w     = tid >> 5;          // wave 0..7
    const int l     = tid & 31;          // lane
    const int lg    = l >> 4;            // lane group (0/1)
    const int ln    = l & 15;
    const int blk   = blockIdx.x;        // 0..2047
    const int b     = blk >> 6;          // 64 M-blocks of 32 rows per batch
    const int srow0 = (blk & 63) << 5;   // first of 32 rows

    v8f zero = {};
    v8f acc0[8], acc1[8];
    #pragma unroll
    for (int i = 0; i < 8; ++i) { acc0[i] = zero; acc1[i] = zero; }

    // Per-lane base of this lane's context rows (tile0: rows 0..15, tile1: +16)
    const float* arow0 =
        context + ((size_t)(b * S_) + srow0 + ln) * D_ + (lg << 3);
    const float* arow1 = arow0 + 16 * D_;

    for (int kt = 0; kt < KT_; ++kt) {
        const float* ap0 = arow0 + kt * 32;
        const float* ap1 = arow1 + kt * 32;
        if (kt + 1 < KT_) {
            __builtin_prefetch(ap0 + 32, 0, 1);
            __builtin_prefetch(ap1 + 32, 0, 1);
        }
        v16bf a0 = load_a_frag(ap0);
        v16bf a1 = load_a_frag(ap1);

        #pragma unroll
        for (int ntl = 0; ntl < 8; ++ntl) {
            int nt = (w << 3) + ntl;
            const v16bf bf = *(const v16bf*)(
                whf + (((size_t)kt * NT_ + nt) * 32 + l) * 16);
            acc0[ntl] = __builtin_amdgcn_wmma_f32_16x16x32_bf16(
                false, a0, false, bf, (short)0, acc0[ntl], false, false);
            acc1[ntl] = __builtin_amdgcn_wmma_f32_16x16x32_bf16(
                false, a1, false, bf, (short)0, acc1[ntl], false, false);
        }
    }

    // Epilogue: fuse bias/coverage/tanh/v-dot.
    // C tile layout: vgpr r, lane l -> M = r + 8*lg, N = (l&15).
    const float inv = invcov[b];
    float covn0[8], covn1[8];
    #pragma unroll
    for (int r = 0; r < 8; ++r) {
        int row0 = srow0 + (lg << 3) + r;
        covn0[r] = (coverage[b * S_ + row0] + 0.001f) * inv;
        covn1[r] = (coverage[b * S_ + row0 + 16] + 0.001f) * inv;
    }

    float contrib0[8], contrib1[8];
    #pragma unroll
    for (int r = 0; r < 8; ++r) { contrib0[r] = 0.f; contrib1[r] = 0.f; }

    #pragma unroll
    for (int ntl = 0; ntl < 8; ++ntl) {
        int n      = (w << 7) + (ntl << 4) + ln;
        float wsum = ws2[b * D_ + n];
        float wc   = Wcov[n];
        float vv   = vw[n];
        #pragma unroll
        for (int r = 0; r < 8; ++r) {
            float v0 = acc0[ntl][r] + wsum + covn0[r] * wc;
            float v1 = acc1[ntl][r] + wsum + covn1[r] * wc;
            contrib0[r] += tanhf(v0) * vv;
            contrib1[r] += tanhf(v1) * vv;
        }
    }

    // Reduce over the 16 lanes sharing a row group, then across waves via LDS.
    #pragma unroll
    for (int r = 0; r < 8; ++r) {
        float v0 = contrib0[r];
        float v1 = contrib1[r];
        v0 += __shfl_xor(v0, 1, 32);  v1 += __shfl_xor(v1, 1, 32);
        v0 += __shfl_xor(v0, 2, 32);  v1 += __shfl_xor(v1, 2, 32);
        v0 += __shfl_xor(v0, 4, 32);  v1 += __shfl_xor(v1, 4, 32);
        v0 += __shfl_xor(v0, 8, 32);  v1 += __shfl_xor(v1, 8, 32);
        contrib0[r] = v0;
        contrib1[r] = v1;
    }

    __shared__ float part[8][32];
    if (ln == 0) {
        #pragma unroll
        for (int r = 0; r < 8; ++r) {
            part[w][(lg << 3) + r]      = contrib0[r];
            part[w][16 + (lg << 3) + r] = contrib1[r];
        }
    }
    __syncthreads();

    if (tid < 32) {
        float s = vb[0];
        #pragma unroll
        for (int ww = 0; ww < 8; ++ww) s += part[ww][tid];
        e_out[b * S_ + srow0 + tid] = s;
    }
}

// ---------------------------------------------------------------------------
// Kernel 5: masked softmax with renormalization.
// softmax is shift-invariant, double max-subtract is idempotent, and the
// mask+renorm folds to: attn = exp(e-max)*mask / sum(exp(e-max)*mask)
// ---------------------------------------------------------------------------
__global__ __launch_bounds__(256) void softmax_kernel(
    const float* __restrict__ e, const int* __restrict__ mask,
    float* __restrict__ attn)
{
    int b = blockIdx.x, tid = threadIdx.x;
    float vals[8];
    float lmax = -1e30f;
    #pragma unroll
    for (int j = 0; j < 8; ++j) {
        vals[j] = e[b * S_ + tid + j * 256];
        lmax = fmaxf(lmax, vals[j]);
    }
    __shared__ float red[256];
    red[tid] = lmax; __syncthreads();
    for (int st = 128; st > 0; st >>= 1) {
        if (tid < st) red[tid] = fmaxf(red[tid], red[tid + st]);
        __syncthreads();
    }
    float m = red[0];
    __syncthreads();

    float ex[8];
    float ls = 0.f;
    #pragma unroll
    for (int j = 0; j < 8; ++j) {
        float mk = (float)mask[b * S_ + tid + j * 256];
        ex[j] = __expf(vals[j] - m) * mk;
        ls += ex[j];
    }
    red[tid] = ls; __syncthreads();
    for (int st = 128; st > 0; st >>= 1) {
        if (tid < st) red[tid] += red[tid + st];
        __syncthreads();
    }
    float rinv = 1.0f / red[0];
    #pragma unroll
    for (int j = 0; j < 8; ++j)
        attn[b * S_ + tid + j * 256] = ex[j] * rinv;
}

// ---------------------------------------------------------------------------
// Kernel 6: partial attn_h over an S-chunk of 512:
//   partial[c][b][d] = sum_{s in chunk c} attn[b,s] * context[b,s,d]
// 4 S-chunks x 16 D-chunks x 32 batches = 2048 blocks to saturate HBM.
// Deterministic: fixed summation order, fixed per-chunk ownership.
// ---------------------------------------------------------------------------
#define SCHUNK_ 512
__global__ __launch_bounds__(64) void weighted_sum_partial_kernel(
    const float* __restrict__ attn, const float* __restrict__ context,
    float* __restrict__ partial)
{
    __shared__ float arow[SCHUNK_];
    int blk    = blockIdx.x;          // ((b*16 + chunkD)*4 + chunkS)
    int chunkS = blk & 3;
    int chunkD = (blk >> 2) & 15;
    int b      = blk >> 6;
    int d      = (chunkD << 6) + threadIdx.x;
    int s0     = chunkS * SCHUNK_;

    for (int i = threadIdx.x; i < SCHUNK_; i += 64)
        arow[i] = attn[b * S_ + s0 + i];
    __syncthreads();

    float acc = 0.f;
    const float* ctx = context + ((size_t)b * S_ + s0) * D_ + d;
    #pragma unroll 8
    for (int s = 0; s < SCHUNK_; ++s) acc += arow[s] * ctx[(size_t)s * D_];
    partial[((size_t)chunkS * B_ + b) * D_ + d] = acc;
}

// Kernel 7: attn_h[b,d] = sum over the 4 S-chunk partials
__global__ __launch_bounds__(256) void reduce_partial_kernel(
    const float* __restrict__ partial, float* __restrict__ out)
{
    int i = blockIdx.x * 256 + threadIdx.x;   // 0 .. B_*D_-1
    out[i] = ((partial[i] + partial[(size_t)B_ * D_ + i]) +
              (partial[2 * (size_t)B_ * D_ + i] + partial[3 * (size_t)B_ * D_ + i]));
}

// ---------------------------------------------------------------------------
extern "C" void kernel_launch(void* const* d_in, const int* in_sizes, int n_in,
                              void* d_out, int out_size, void* d_ws, size_t ws_size,
                              hipStream_t stream)
{
    const float* state_h  = (const float*)d_in[0];
    const float* state_c  = (const float*)d_in[1];
    const float* context  = (const float*)d_in[2];
    const int*   mask     = (const int*)  d_in[3];
    /* d_in[4] = last_attn: unused by the reference */
    const float* coverage = (const float*)d_in[5];
    const float* Wh       = (const float*)d_in[6];
    const float* bh       = (const float*)d_in[7];
    const float* Ws       = (const float*)d_in[8];
    const float* bs       = (const float*)d_in[9];
    const float* Wcov     = (const float*)d_in[10];
    const float* bcov     = (const float*)d_in[11];
    const float* vw       = (const float*)d_in[12];
    const float* vb       = (const float*)d_in[13];

    // Workspace layout (all 256B aligned): ~2.9 MB total
    char*   ws      = (char*)d_ws;
    __bf16* frag    = (__bf16*)ws;                              // 2 MB
    size_t  off     = (size_t)D_ * D_ * sizeof(__bf16);
    float*  ws2     = (float*)(ws + off);  off += (size_t)B_ * D_ * sizeof(float);
    float*  invc    = (float*)(ws + off);  off += 256;
    float*  e       = (float*)(ws + off);  off += (size_t)B_ * S_ * sizeof(float);
    float*  partial = (float*)(ws + off);  // 4 * B_ * D_ floats

    float* attn_h = (float*)d_out;            // (B, D)
    float* attn   = (float*)d_out + B_ * D_;  // (B, S)

    swizzle_wh_kernel           <<<4096, 256, 0, stream>>>(Wh, frag);
    wstate_kernel               <<<B_ * 16, 64, 0, stream>>>(state_h, state_c, Ws, bs, bh, bcov, ws2);
    covsum_kernel               <<<B_, 256, 0, stream>>>(coverage, invc);
    attn_e_kernel               <<<(B_ * S_) / 32, 256, 0, stream>>>(
                                    context, coverage, frag, ws2, invc, Wcov, vw, vb, e);
    softmax_kernel              <<<B_, 256, 0, stream>>>(e, mask, attn);
    weighted_sum_partial_kernel <<<B_ * 16 * 4, 64, 0, stream>>>(attn, context, partial);
    reduce_partial_kernel       <<<(B_ * D_) / 256, 256, 0, stream>>>(partial, attn_h);
}
